// BigramHash_16810501996721
// MI455X (gfx1250) — compile-verified, matrix operand order checked by reading, not run on hardware
//
#include <hip/hip_runtime.h>

typedef _Float16 v16h __attribute__((ext_vector_type(16)));
typedef _Float16 v8h  __attribute__((ext_vector_type(8)));
typedef float    v8f  __attribute__((ext_vector_type(8)));

#define BUCKETS   8192
#define DB        48
#define DIM       512
#define NTOK      (32 * 4096)
#define W_STRIDE  72          // fp16 K-stride in LDS: 144B = 36 banks -> conflict-free b128;
                              // K in [48,72) is zero so the b1 fragment's upper half loads as zeros
#define BLOCK     256

__global__ __launch_bounds__(BLOCK) void bigram_proj_kernel(
    const int* __restrict__ idx,
    const _Float16* __restrict__ table,
    const float* __restrict__ projw,
    float* __restrict__ out)
{
    // proj_w [512][48] f32 -> LDS f16, K zero-padded through the stride
    __shared__ _Float16 Wlds[DIM * W_STRIDE];   // 72 KB (<320 KB/WGP)
    for (int e = threadIdx.x; e < DIM * W_STRIDE; e += BLOCK) {
        int n = e / W_STRIDE;
        int k = e - n * W_STRIDE;
        _Float16 v = (_Float16)0.f;
        if (k < DB) v = (_Float16)projw[n * DB + k];
        Wlds[e] = v;
    }
    __syncthreads();

    const int lane = threadIdx.x & 31;
    const int wave = threadIdx.x >> 5;
    const int m    = lane & 15;     // row-within-halfwave: A's M, B's N, D's N
    const int hs   = lane >> 4;     // half-select (K-interleave / D row offset)
    const int hoff = hs * 8;        // fp16-element offset for this half's K group

    // each wave: two 16-token strips
    const int t0 = (blockIdx.x * 8 + wave) * 32;

    // ---- gather A fragments (hashed bigram rows of the fp16 table) ----
    v16h a0[2], a1[2];
#pragma unroll
    for (int r = 0; r < 2; ++r) {
        const int tok = t0 + r * 16 + m;
        const int s = tok & 4095;            // S = 4096
        int bucket = 0;
        if (s != 0) {
            int i0 = idx[tok - 1];
            int i1 = idx[tok];
            bucket = (i0 * 10007 + i1) & (BUCKETS - 1);
        }
        const _Float16* row = table + bucket * DB;        // 96B row, 16B aligned
        v8h lo0 = *(const v8h*)(row + hoff);              // K 0..7   | 8..15
        v8h hi0 = *(const v8h*)(row + 16 + hoff);         // K 16..23 | 24..31
        v8h lo1 = *(const v8h*)(row + 32 + hoff);         // K 32..39 | 40..47
#pragma unroll
        for (int i = 0; i < 8; ++i) {
            a0[r][i]     = lo0[i];
            a0[r][8 + i] = hi0[i];
            a1[r][i]     = lo1[i];
            a1[r][8 + i] = (_Float16)0.f;                 // K 48..63 = padding (hoisted, one-time)
        }
    }

    // ---- sweep all 32 N-tiles of DIM=512 ----
#pragma unroll 4
    for (int nt = 0; nt < DIM / 16; ++nt) {
        const int n = nt * 16 + m;
        const _Float16* wrow = &Wlds[n * W_STRIDE + hoff];
        // all four halves are pure LDS loads -> coalesce into ds_load_b128 pairs
        v8h blo0 = *(const v8h*)(wrow);                   // K 0..7   | 8..15
        v8h bhi0 = *(const v8h*)(wrow + 16);              // K 16..23 | 24..31
        v8h blo1 = *(const v8h*)(wrow + 32);              // K 32..39 | 40..47
        v8h bhi1 = *(const v8h*)(wrow + 48);              // K 48..55 | 56..63 (zeros in LDS)
        v16h b0, b1;
#pragma unroll
        for (int i = 0; i < 8; ++i) {
            b0[i]     = blo0[i];
            b0[8 + i] = bhi0[i];
            b1[i]     = blo1[i];
            b1[8 + i] = bhi1[i];
        }

#pragma unroll
        for (int r = 0; r < 2; ++r) {
            v8f c = {};
            c = __builtin_amdgcn_wmma_f32_16x16x32_f16(false, a0[r], false, b0,
                                                       (short)0, c, false, false);
            c = __builtin_amdgcn_wmma_f32_16x16x32_f16(false, a1[r], false, b1,
                                                       (short)0, c, false, false);
            // D layout: VGPR v -> row M = hs*8 + v, lane -> col N = m
            float* obase = out + (size_t)(t0 + r * 16 + hs * 8) * DIM + nt * 16 + m;
#pragma unroll
            for (int v = 0; v < 8; ++v)
                __builtin_nontemporal_store(c[v], obase + (size_t)v * DIM);
        }
    }
}

extern "C" void kernel_launch(void* const* d_in, const int* in_sizes, int n_in,
                              void* d_out, int out_size, void* d_ws, size_t ws_size,
                              hipStream_t stream) {
    (void)in_sizes; (void)n_in; (void)out_size; (void)d_ws; (void)ws_size;
    const int*      idx   = (const int*)d_in[0];
    const _Float16* table = (const _Float16*)d_in[1];
    const float*    projw = (const float*)d_in[2];
    float*          out   = (float*)d_out;

    // 131072 tokens / 256 tokens-per-block = 512 blocks, 256 threads (8 wave32)
    bigram_proj_kernel<<<dim3(NTOK / 256), dim3(BLOCK), 0, stream>>>(idx, table, projw, out);
}
